// MultiHaedAttention_75144747811336
// MI455X (gfx1250) — matrix operand order repaired and checked by reading, not compiled, not run on hardware
//
#include <hip/hip_runtime.h>

// ---------------------------------------------------------------------------
// MultiHeadAttention forward for MI455X (gfx1250), bf16 WMMA path.
// out = softmax((xWq)(xWk)^T * d^-0.5) (xWv) Wout + b
// B=8, N=1024, DIM=128, HEADS=8, per-head dim=128, inner=1024.
// ---------------------------------------------------------------------------

typedef unsigned short ushort_t;
typedef __attribute__((ext_vector_type(16))) __bf16 v16bf;
typedef __attribute__((ext_vector_type(8)))  float  v8f;
typedef int v4i_vs __attribute__((vector_size(16)));   // matches builtin param

#define BATCH 8
#define SEQ   1024
#define DM    128
#define HEADS 8
#define INNER 1024              // HEADS * DM
#define ROWS  (BATCH * SEQ)     // 8192
#define QKVN  (3 * INNER)       // 3072
#define KSTEP 32
#define NSTEPS (SEQ / KSTEP)    // 32
#define QK_SCALE 0.08838834764831845f   // 128^-0.5

// ---- gfx1250 async global->LDS copy (ASYNCcnt) with safe fallback ---------
#if defined(__has_builtin)
#  if __has_builtin(__builtin_amdgcn_global_load_async_to_lds_b128)
#    define HAS_ASYNC_LDS 1
#  endif
#  if __has_builtin(__builtin_amdgcn_s_wait_asynccnt)
#    define HAS_ASYNC_WAIT 1
#  endif
#endif
#ifndef HAS_ASYNC_LDS
#  define HAS_ASYNC_LDS 0
#endif
#ifndef HAS_ASYNC_WAIT
#  define HAS_ASYNC_WAIT 0
#endif

union FragU { v16bf v; uint4 q[2]; };

__device__ __forceinline__ void async_b128(const ushort_t* g, ushort_t* l) {
#if HAS_ASYNC_LDS
  __builtin_amdgcn_global_load_async_to_lds_b128(
      (__attribute__((address_space(1))) v4i_vs*)g,
      (__attribute__((address_space(3))) v4i_vs*)l, 0, 0);
#else
  *(uint4*)l = *(const uint4*)g;      // synchronous fallback
#endif
}
__device__ __forceinline__ void async_wait0() {
#if HAS_ASYNC_WAIT
  __builtin_amdgcn_s_wait_asynccnt(0);
#endif
}

// f32 -> bf16 round-to-nearest-even (bit trick, branch-free)
__device__ __forceinline__ ushort_t f2bf(float f) {
  unsigned u = __builtin_bit_cast(unsigned, f);
  u += 0x7FFFu + ((u >> 16) & 1u);
  return (ushort_t)(u >> 16);
}

// A-fragment (16x32 bf16). rowPtr points at row (lane&15), kBase.
__device__ __forceinline__ v16bf load_a(const ushort_t* rowPtr, int half) {
  FragU u;
  const ushort_t* p = rowPtr + half * 8;
  u.q[0] = *(const uint4*)(p);
  u.q[1] = *(const uint4*)(p + 16);
  return u.v;
}

// B-fragment (32x16 bf16) from src laid out [n][k] row-major (stride elems).
__device__ __forceinline__ v16bf load_b(const ushort_t* src, int stride,
                                        int colBase, int kBase, int lane) {
  FragU u;
  const ushort_t* p = src + (size_t)(colBase + (lane & 15)) * stride
                          + kBase + (lane >> 4) * 8;
  u.q[0] = *(const uint4*)(p);
  u.q[1] = *(const uint4*)(p + 16);
  return u.v;
}

__device__ __forceinline__ v8f wmma_bf16(v16bf a, v16bf b, v8f c) {
  return __builtin_amdgcn_wmma_f32_16x16x32_bf16(
      false, a, false, b, (short)0, c, false, false);
}

// ---------------------------------------------------------------------------
// Kernel 0: f32 -> bf16 conversions; weights transposed to [n][k].
// ---------------------------------------------------------------------------
__global__ void convert_kernel(const float* __restrict__ x,
                               const float* __restrict__ wqkv,
                               const float* __restrict__ wout,
                               ushort_t* __restrict__ xb,
                               ushort_t* __restrict__ wqkvt,
                               ushort_t* __restrict__ woutt) {
  int stride = gridDim.x * blockDim.x;
  int tid = blockIdx.x * blockDim.x + threadIdx.x;
  for (int i = tid; i < ROWS * DM; i += stride)
    xb[i] = f2bf(x[i]);
  for (int i = tid; i < QKVN * DM; i += stride) {   // wqkvt[n][k] = wqkv[k][n]
    int n = i / DM, k = i % DM;
    wqkvt[i] = f2bf(wqkv[(size_t)k * QKVN + n]);
  }
  for (int i = tid; i < DM * INNER; i += stride) {  // woutt[n][k] = wout[k][n]
    int n = i / INNER, k = i % INNER;
    woutt[i] = f2bf(wout[(size_t)k * DM + n]);
  }
}

// ---------------------------------------------------------------------------
// Kernel 1: QKV GEMM, wave = 16x64 strip (A-fragment reused across 4 WMMAs).
// Routes into Q (pre-scaled by d^-0.5), K [j][d], V^T [d][j], all bf16.
// ---------------------------------------------------------------------------
__global__ __launch_bounds__(256) void qkv_gemm_kernel(
    const ushort_t* __restrict__ xb, const ushort_t* __restrict__ wqkvt,
    ushort_t* __restrict__ Qb, ushort_t* __restrict__ Kb,
    ushort_t* __restrict__ Vtb) {
  int lane = threadIdx.x & 31, wv = threadIdx.x >> 5;
  int half = lane >> 4, ln = lane & 15;
  int tile = blockIdx.x * 8 + wv;          // 512 * 48 = 24576 waves
  int mt = tile / (QKVN / 64);
  int n64 = tile % (QKVN / 64);

  v8f acc[4];
#pragma unroll
  for (int s = 0; s < 4; ++s) { v8f z = {}; acc[s] = z; }

#pragma unroll
  for (int kk = 0; kk < 4; ++kk) {
    v16bf a = load_a(xb + (size_t)(mt * 16 + ln) * DM + kk * 32, half);
#pragma unroll
    for (int s = 0; s < 4; ++s) {
      v16bf b = load_b(wqkvt, DM, n64 * 64 + s * 16, kk * 32, lane);
      acc[s] = wmma_bf16(a, b, acc[s]);
    }
  }

  int rowBase = mt * 16;
  int b = rowBase / SEQ;          // uniform
  int ibase = rowBase % SEQ;
#pragma unroll
  for (int s = 0; s < 4; ++s) {
    int n0 = n64 * 64 + s * 16;
    int sector = n0 / INNER;      // uniform: 0=Q 1=K 2=V
    int nin = n0 % INNER;
    int h = nin / DM;             // uniform (16-wide tile inside one head)
    int dcol = (nin % DM) + ln;
    size_t headOff = (size_t)b * HEADS + h;
    if (sector == 0) {
#pragma unroll
      for (int r = 0; r < 8; ++r) {
        int i = ibase + r + 8 * half;
        Qb[(headOff * SEQ + i) * DM + dcol] = f2bf(acc[s][r] * QK_SCALE);
      }
    } else if (sector == 1) {
#pragma unroll
      for (int r = 0; r < 8; ++r) {
        int i = ibase + r + 8 * half;
        Kb[(headOff * SEQ + i) * DM + dcol] = f2bf(acc[s][r]);
      }
    } else {
#pragma unroll
      for (int r = 0; r < 8; ++r) {
        int i = ibase + r + 8 * half;
        Vtb[(headOff * DM + dcol) * SEQ + i] = f2bf(acc[s][r]);  // transposed
      }
    }
  }
}

// ---------------------------------------------------------------------------
// Kernel 2: flash attention. Block = 8 waves covering 128 query rows of one
// (b,h). K/V tiles double-buffered in LDS via async global->LDS copies; all
// 8 waves share each staged tile (8x less global traffic). Per 32-key step:
// 8 score WMMAs, online softmax (shfl_xor over 16-lane halves), P re-layout
// through per-wave LDS (C-frag -> A-frag), 8 P*V WMMAs.
// ---------------------------------------------------------------------------
__global__ __launch_bounds__(256) void attn_kernel(
    const ushort_t* __restrict__ Qb, const ushort_t* __restrict__ Kb,
    const ushort_t* __restrict__ Vtb, ushort_t* __restrict__ zb) {
  __shared__ __align__(16) ushort_t ksm[2][KSTEP * DM];   // [j][d], 8KB each
  __shared__ __align__(16) ushort_t vsm[2][DM * KSTEP];   // [d][j], 8KB each
  __shared__ __align__(16) ushort_t psm[8 * 512];         // 16x32 P per wave

  int tid = threadIdx.x;
  int lane = tid & 31, wv = tid >> 5;
  int half = lane >> 4, ln = lane & 15;
  int wg = blockIdx.x;
  int b = wg >> 6, rem = wg & 63;
  int h = rem >> 3, blk = rem & 7;
  int i0 = blk * 128 + wv * 16;

  const ushort_t* Qh = Qb + ((size_t)b * HEADS + h) * SEQ * DM;
  const ushort_t* Kh = Kb + ((size_t)b * HEADS + h) * SEQ * DM;
  const ushort_t* Vh = Vtb + ((size_t)b * HEADS + h) * DM * SEQ;
  ushort_t* pw = psm + wv * 512;

  // stage one 32-key K tile + V^T tile: 4 async b128 chunks per thread
  auto stage = [&](int kc, int buf) {
#pragma unroll
    for (int c = 0; c < 2; ++c) {
      int e = (tid + c * 256) * 8;          // flat bf16 element index
      int krow = e >> 7, kcol = e & 127;    // ksm: [32][128]
      async_b128(Kh + (size_t)(kc + krow) * DM + kcol, &ksm[buf][e]);
      int dr = e >> 5, j = e & 31;          // vsm: [128][32]
      async_b128(Vh + (size_t)dr * SEQ + kc + j, &vsm[buf][e]);
    }
  };

  v16bf qa[4];                              // Q tile resident: 16x128
#pragma unroll
  for (int kk = 0; kk < 4; ++kk)
    qa[kk] = load_a(Qh + (size_t)(i0 + ln) * DM + kk * 32, half);

  v8f acc[8];
#pragma unroll
  for (int t = 0; t < 8; ++t) { v8f z = {}; acc[t] = z; }
  float mrow[8], lrow[8];
#pragma unroll
  for (int r = 0; r < 8; ++r) { mrow[r] = -3.4e38f; lrow[r] = 0.f; }

  stage(0, 0);
  async_wait0();
  __syncthreads();

  for (int step = 0; step < NSTEPS; ++step) {
    int buf = step & 1;
    if (step + 1 < NSTEPS) stage((step + 1) * KSTEP, buf ^ 1);  // overlap

    const ushort_t* kt = ksm[buf];
    const ushort_t* vt = vsm[buf];

    // ---- scores S = Q K^T (Q pre-scaled); two 16-col tiles, B from LDS ----
    v8f s0 = {}, s1 = {};
#pragma unroll
    for (int kk = 0; kk < 4; ++kk) {
      v16bf kb0 = load_b(kt, DM, 0, kk * 32, lane);
      v16bf kb1 = load_b(kt, DM, 16, kk * 32, lane);
      s0 = wmma_bf16(qa[kk], kb0, s0);
      s1 = wmma_bf16(qa[kk], kb1, s1);
    }

    // ---- online softmax per row (row = VGPR r + 8*half, cols over lanes) --
    float corr[8];
#pragma unroll
    for (int r = 0; r < 8; ++r) {
      float mx = fmaxf(s0[r], s1[r]);
      mx = fmaxf(mx, __shfl_xor(mx, 1, 32));
      mx = fmaxf(mx, __shfl_xor(mx, 2, 32));
      mx = fmaxf(mx, __shfl_xor(mx, 4, 32));
      mx = fmaxf(mx, __shfl_xor(mx, 8, 32));
      float mnew = fmaxf(mrow[r], mx);
      corr[r] = __expf(mrow[r] - mnew);
      float p0 = __expf(s0[r] - mnew);
      float p1 = __expf(s1[r] - mnew);
      float sum = p0 + p1;
      sum += __shfl_xor(sum, 1, 32);
      sum += __shfl_xor(sum, 2, 32);
      sum += __shfl_xor(sum, 4, 32);
      sum += __shfl_xor(sum, 8, 32);
      lrow[r] = lrow[r] * corr[r] + sum;
      mrow[r] = mnew;
      int row = r + 8 * half;
      pw[row * 32 + ln] = f2bf(p0);         // C-frag layout -> LDS [row][col]
      pw[row * 32 + 16 + ln] = f2bf(p1);
    }
#pragma unroll
    for (int t = 0; t < 8; ++t)
#pragma unroll
      for (int r = 0; r < 8; ++r) acc[t][r] *= corr[r];

    // ---- reload P as A-fragment (DS ops in-order within wave) ----
    FragU pu;
    const ushort_t* pp = pw + ln * 32 + half * 8;
    pu.q[0] = *(const uint4*)(pp);
    pu.q[1] = *(const uint4*)(pp + 16);
    v16bf pa = pu.v;

    // ---- z += P * V  (V^T tile in LDS => contiguous B-fragments) ----
#pragma unroll
    for (int t = 0; t < 8; ++t) {
      v16bf vb = load_b(vt, KSTEP, t * 16, 0, lane);
      acc[t] = wmma_bf16(pa, vb, acc[t]);
    }

    async_wait0();      // next tile's copies have landed (this wave)
    __syncthreads();    // ...and everyone else's; safe to swap buffers
  }

  // ---- normalize and emit z in [B*N][inner] bf16 (A-frag friendly) ----
#pragma unroll
  for (int r = 0; r < 8; ++r) {
    float inv = 1.f / lrow[r];
    int i = i0 + r + 8 * half;
    size_t rowOff = ((size_t)b * SEQ + i) * INNER + (size_t)h * DM;
#pragma unroll
    for (int t = 0; t < 8; ++t)
      zb[rowOff + t * 16 + ln] = f2bf(acc[t][r] * inv);
  }
}

// ---------------------------------------------------------------------------
// Kernel 3: output projection, wave = 16x64 strip, 128 WMMAs over K=1024.
// ---------------------------------------------------------------------------
__global__ __launch_bounds__(256) void out_gemm_kernel(
    const ushort_t* __restrict__ zb, const ushort_t* __restrict__ woutt,
    const float* __restrict__ bo, float* __restrict__ out) {
  int lane = threadIdx.x & 31, wv = threadIdx.x >> 5;
  int half = lane >> 4, ln = lane & 15;
  int tile = blockIdx.x * 8 + wv;          // 512 * 2 = 1024 waves
  int mt = tile >> 1, n64 = tile & 1;

  v8f acc[4];
#pragma unroll
  for (int s = 0; s < 4; ++s) { v8f z = {}; acc[s] = z; }

#pragma unroll 4
  for (int kk = 0; kk < 32; ++kk) {
    v16bf a = load_a(zb + (size_t)(mt * 16 + ln) * INNER + kk * 32, half);
#pragma unroll
    for (int s = 0; s < 4; ++s) {
      v16bf b = load_b(woutt, INNER, n64 * 64 + s * 16, kk * 32, lane);
      acc[s] = wmma_bf16(a, b, acc[s]);
    }
  }
#pragma unroll
  for (int s = 0; s < 4; ++s) {
    int n0 = n64 * 64 + s * 16;
    float bias = bo[n0 + ln];
#pragma unroll
    for (int r = 0; r < 8; ++r) {
      int row = mt * 16 + r + 8 * half;
      out[(size_t)row * DM + n0 + ln] = acc[s][r] + bias;
    }
  }
}

// ---------------------------------------------------------------------------
extern "C" void kernel_launch(void* const* d_in, const int* in_sizes, int n_in,
                              void* d_out, int out_size, void* d_ws, size_t ws_size,
                              hipStream_t stream) {
  const float* x    = (const float*)d_in[0];   // [8,1024,128]
  const float* wqkv = (const float*)d_in[1];   // [128,3072]
  const float* wout = (const float*)d_in[2];   // [1024,128]
  const float* bo   = (const float*)d_in[3];   // [128]
  float* out = (float*)d_out;                  // [8,1024,128]

  // workspace carve-up (bf16 = ushort), ~67 MB total
  ushort_t* xb    = (ushort_t*)d_ws;
  ushort_t* wqkvt = xb    + (size_t)ROWS * DM;
  ushort_t* woutt = wqkvt + (size_t)QKVN * DM;
  ushort_t* Qb    = woutt + (size_t)DM * INNER;
  ushort_t* Kb    = Qb    + (size_t)BATCH * HEADS * SEQ * DM;
  ushort_t* Vtb   = Kb    + (size_t)BATCH * HEADS * SEQ * DM;
  ushort_t* zb    = Vtb   + (size_t)BATCH * HEADS * SEQ * DM;

  convert_kernel<<<1024, 256, 0, stream>>>(x, wqkv, wout, xb, wqkvt, woutt);

  // 512 M-tiles * 48 N64-strips = 24576 waves / 8 per block
  qkv_gemm_kernel<<<24576 / 8, 256, 0, stream>>>(xb, wqkvt, Qb, Kb, Vtb);

  // B*H*(N/128) blocks; each block's 8 waves cover 128 query rows
  attn_kernel<<<BATCH * HEADS * (SEQ / 128), 256, 0, stream>>>(Qb, Kb, Vtb, zb);

  // 512 M-tiles * 2 N64-strips = 1024 waves / 8 per block
  out_gemm_kernel<<<1024 / 8, 256, 0, stream>>>(zb, woutt, bo, out);
}
